// ParallelExperts_71579924955712
// MI455X (gfx1250) — matrix operand order repaired and checked by reading, not compiled, not run on hardware
//
#include <hip/hip_runtime.h>

// Problem constants (from the reference)
constexpr int kE   = 64;
constexpr int kIN  = 512;
constexpr int kOUT = 512;
constexpr int kCAP = 3072;

// Tiling
constexpr int BM = 128;
constexpr int BN = 128;
constexpr int BK = 32;
constexpr int LDR = 40;        // bf16 elements per LDS row (32 + 8 pad) = 80 bytes

typedef float  v8f   __attribute__((ext_vector_type(8)));
typedef __bf16 v4bf  __attribute__((ext_vector_type(4)));
typedef __bf16 v8bf  __attribute__((ext_vector_type(8)));
typedef __bf16 v16bf __attribute__((ext_vector_type(16)));

// Build a 16-element bf16 fragment from two 16-byte LDS groups (ds_load_b128 x2)
__device__ __forceinline__ v16bf frag16(const __bf16* p0, const __bf16* p1) {
    v8bf g0 = *(const v8bf*)p0;
    v8bf g1 = *(const v8bf*)p1;
    return __builtin_shufflevector(g0, g1, 0,1,2,3,4,5,6,7,8,9,10,11,12,13,14,15);
}

__global__ __launch_bounds__(256)
void moe_grouped_gemm_bf16x3(const float* __restrict__ x,    // [N, IN]
                             const int*   __restrict__ esz,  // [E]
                             const float* __restrict__ w,    // [E, IN, OUT]
                             float*       __restrict__ out)  // [N, OUT]
{
    // hi/lo split planes; A in [m][k] order, B transposed to [n][k] order
    __shared__ __bf16 a_hi[BM * LDR];
    __shared__ __bf16 a_lo[BM * LDR];
    __shared__ __bf16 b_hi[BN * LDR];
    __shared__ __bf16 b_lo[BN * LDR];

    const int e     = blockIdx.z;
    const int esize = esz[e];
    const int bm0   = blockIdx.y * BM;
    if (bm0 >= esize) return;                 // capacity padding: skip empty tiles
    const int bn0   = blockIdx.x * BN;

    // uniform prefix sum -> scalar loads
    int offset = 0;
    for (int i = 0; i < e; ++i) offset += esz[i];

    const int tid  = threadIdx.x;
    const int lane = tid & 31;
    const int wave = tid >> 5;
    const int wm   = wave >> 1;   // 0..3: 32-row strip
    const int wn   = wave & 1;    // 0..1: 64-col strip
    const int hi   = lane >> 4;   // half-wave
    const int l16  = lane & 15;

    const int mrows = min(BM, esize - bm0);

    const float* Ag = x + (size_t)(offset + bm0) * kIN;
    const float* Wg = w + (size_t)e * kIN * kOUT + bn0;

    v8f acc[2][4] = {};

    // ---- staging registers (software pipeline: loads of chunk c+1 fly under compute of c)
    float4 aReg[4];        // A: (m = id>>3, k-quad q = id&7)
    float  bReg[4][4];     // B: (n = id&127, k-quad g = id>>7), 4 k per task

    auto load_chunk = [&](int k0) {
        #pragma unroll
        for (int i = 0; i < 4; ++i) {
            const int id = i * 256 + tid;
            const int m  = id >> 3;
            const int q  = id & 7;
            const int gm = (m < mrows) ? m : 0;       // clamp: garbage rows never stored
            aReg[i] = *(const float4*)(Ag + (size_t)gm * kIN + k0 + q * 4);
        }
        #pragma unroll
        for (int i = 0; i < 4; ++i) {
            const int id = i * 256 + tid;
            const int n  = id & 127;                  // consecutive lanes -> consecutive n
            const int g  = id >> 7;
            #pragma unroll
            for (int kk = 0; kk < 4; ++kk)
                bReg[i][kk] = Wg[(size_t)(k0 + g * 4 + kk) * kOUT + n];
        }
    };

    auto split_store = [&]() {
        #pragma unroll
        for (int i = 0; i < 4; ++i) {
            const int id = i * 256 + tid;
            const int m  = id >> 3;
            const int q  = id & 7;
            const float* pv = (const float*)&aReg[i];
            v4bf h, l;
            #pragma unroll
            for (int kk = 0; kk < 4; ++kk) {
                const float  xv = pv[kk];
                const __bf16 hh = (__bf16)xv;
                h[kk] = hh;
                l[kk] = (__bf16)(xv - (float)hh);
            }
            *(v4bf*)(&a_hi[m * LDR + q * 4]) = h;
            *(v4bf*)(&a_lo[m * LDR + q * 4]) = l;
        }
        #pragma unroll
        for (int i = 0; i < 4; ++i) {
            const int id = i * 256 + tid;
            const int n  = id & 127;
            const int g  = id >> 7;
            v4bf h, l;
            #pragma unroll
            for (int kk = 0; kk < 4; ++kk) {
                const float  xv = bReg[i][kk];
                const __bf16 hh = (__bf16)xv;
                h[kk] = hh;
                l[kk] = (__bf16)(xv - (float)hh);
            }
            *(v4bf*)(&b_hi[n * LDR + g * 4]) = h;   // transposed: row n, k-quad g
            *(v4bf*)(&b_lo[n * LDR + g * 4]) = l;
        }
    };

    load_chunk(0);

    constexpr int NCHUNK = kIN / BK;   // 16
    for (int c = 0; c < NCHUNK; ++c) {
        __syncthreads();               // previous chunk's LDS reads complete
        split_store();
        __syncthreads();
        if (c + 1 < NCHUNK) load_chunk((c + 1) * BK);   // overlap with WMMA section

        // A fragments: 16-bit A 16x32 map -> lane-half h reads k in [8h,8h+8) u [16+8h,16+8h+8)
        v16bf ah[2], al[2];
        #pragma unroll
        for (int i = 0; i < 2; ++i) {
            const int row = (wm * 32 + i * 16 + l16) * LDR;
            ah[i] = frag16(&a_hi[row + 8 * hi], &a_hi[row + 16 + 8 * hi]);
            al[i] = frag16(&a_lo[row + 8 * hi], &a_lo[row + 16 + 8 * hi]);
        }
        // B fragments: 16-bit B 32x16 map -> lane-half h reads k in [8h,8h+8) u [16+8h,16+8h+8), matching A
        #pragma unroll
        for (int j = 0; j < 4; ++j) {
            const int row = (wn * 64 + j * 16 + l16) * LDR;
            const v16bf bh = frag16(&b_hi[row + 8 * hi], &b_hi[row + 16 + 8 * hi]);
            const v16bf bl = frag16(&b_lo[row + 8 * hi], &b_lo[row + 16 + 8 * hi]);
            #pragma unroll
            for (int i = 0; i < 2; ++i) {
                acc[i][j] = __builtin_amdgcn_wmma_f32_16x16x32_bf16(
                    false, ah[i], false, bh, (short)0, acc[i][j], false, false);
                acc[i][j] = __builtin_amdgcn_wmma_f32_16x16x32_bf16(
                    false, ah[i], false, bl, (short)0, acc[i][j], false, false);
                acc[i][j] = __builtin_amdgcn_wmma_f32_16x16x32_bf16(
                    false, al[i], false, bh, (short)0, acc[i][j], false, false);
            }
        }
    }

    // Epilogue: C layout -> lane holds (m = 8*hi + r, n = l16) of each 16x16 subtile
    float* Og = out + (size_t)(offset + bm0) * kOUT + bn0;
    #pragma unroll
    for (int i = 0; i < 2; ++i) {
        #pragma unroll
        for (int r = 0; r < 8; ++r) {
            const int rl = wm * 32 + i * 16 + hi * 8 + r;
            if (rl < mrows) {
                #pragma unroll
                for (int j = 0; j < 4; ++j)
                    Og[(size_t)rl * kOUT + wn * 64 + j * 16 + l16] = acc[i][j][r];
            }
        }
    }
}

extern "C" void kernel_launch(void* const* d_in, const int* in_sizes, int n_in,
                              void* d_out, int out_size, void* d_ws, size_t ws_size,
                              hipStream_t stream) {
    const float* x   = (const float*)d_in[0];   // inputs  [N, IN] f32
    const int*   esz = (const int*)  d_in[1];   // expert_size [E] i32
    const float* w   = (const float*)d_in[2];   // weight  [E, IN, OUT] f32
    float*       out = (float*)d_out;           // [N, OUT] f32

    dim3 grid(kOUT / BN, kCAP / BM, kE);        // (4, 24, 64)
    moe_grouped_gemm_bf16x3<<<grid, 256, 0, stream>>>(x, esz, w, out);
}
